// GNNMoELayer_11879879544434
// MI455X (gfx1250) — matrix operand order repaired and checked by reading, not compiled, not run on hardware
//
#include <hip/hip_runtime.h>
#include <hip/hip_bf16.h>
#include <math.h>

#define M_TOKENS (16 * 4096)
#define D_DIM    256
#define D2       512
#define NEXP     4
#define NHEADS   4
#define NEDGES   1048576
#define ETOT     (NEDGES + M_TOKENS)

typedef __attribute__((ext_vector_type(16))) __bf16    v16bf;
typedef __attribute__((ext_vector_type(8)))  __bf16    v8bf;
typedef __attribute__((ext_vector_type(8)))  float     v8f;
typedef __attribute__((ext_vector_type(4)))  unsigned  v4u;
typedef __attribute__((ext_vector_type(8)))  int       v8i;
typedef __attribute__((ext_vector_type(4)))  int       v4i;

// ---- monotone uint encoding of float for atomicMax-based segment_max ----
__device__ __forceinline__ unsigned fenc(float f) {
    unsigned u = __float_as_uint(f);
    return (u & 0x80000000u) ? ~u : (u | 0x80000000u);
}
__device__ __forceinline__ float fdec(unsigned u) {
    unsigned b = (u & 0x80000000u) ? (u & 0x7FFFFFFFu) : ~u;
    return __uint_as_float(b);
}

// =====================================================================
// Kernel 0: precision/layout staging.
//   xb   = bf16(x)                       [M,256]
//   W1t  = bf16(W1^T per expert)         [E][512][256]  (col-major of W1)
//   W2t  = bf16(W2^T per expert)         [E][256][512]  (col-major of W2)
// Makes every WMMA B-fragment two contiguous 16B loads.
// =====================================================================
__global__ void convert_kernel(const float* __restrict__ x,
                               const float* __restrict__ W1,
                               const float* __restrict__ W2,
                               __bf16* __restrict__ xb,
                               __bf16* __restrict__ W1t,
                               __bf16* __restrict__ W2t) {
    size_t i = (size_t)blockIdx.x * blockDim.x + threadIdx.x;
    if (i < (size_t)M_TOKENS * D_DIM) xb[i] = (__bf16)x[i];
    if (i < (size_t)NEXP * D_DIM * D2) {
        int ei  = (int)(i / (D_DIM * D2));
        int rem = (int)(i % (D_DIM * D2));
        // W1t[ei][n][k] = W1[ei][k][n]   (n<512, k<256)
        int n1 = rem / D_DIM, k1 = rem % D_DIM;
        W1t[i] = (__bf16)W1[(size_t)ei * D_DIM * D2 + (size_t)k1 * D2 + n1];
        // W2t[ei][n][k] = W2[ei][k][n]   (n<256, k<512)
        int n2 = rem / D2, k2 = rem % D2;
        W2t[i] = (__bf16)W2[(size_t)ei * D2 * D_DIM + (size_t)k2 * D_DIM + n2];
    }
}

// =====================================================================
// Kernel 1: gating projection h = x @ W_gat, per-head attention scalars,
//           and zero-init of segment buffers.
// =====================================================================
__global__ void gating_kernel(const float* __restrict__ x,
                              const float* __restrict__ Wg,
                              const float* __restrict__ att_src,
                              const float* __restrict__ att_dst,
                              float* __restrict__ h,
                              float* __restrict__ a_src,
                              float* __restrict__ a_dst,
                              unsigned* __restrict__ emax,
                              float* __restrict__ denom,
                              float* __restrict__ agg) {
    int r = blockIdx.x * blockDim.x + threadIdx.x;
    if (r >= M_TOKENS) return;
    const float* xr = x + (size_t)r * D_DIM;
    float acc[16];
#pragma unroll
    for (int c = 0; c < 16; ++c) acc[c] = 0.f;
    for (int d = 0; d < D_DIM; ++d) {
        float xv = xr[d];
        const float* wrow = Wg + d * 16;
#pragma unroll
        for (int c = 0; c < 16; ++c) acc[c] += xv * wrow[c];
    }
#pragma unroll
    for (int c = 0; c < 16; ++c) h[(size_t)r * 16 + c] = acc[c];
#pragma unroll
    for (int hd = 0; hd < NHEADS; ++hd) {
        float s = 0.f, dd = 0.f;
#pragma unroll
        for (int c = 0; c < NEXP; ++c) {
            s  += acc[hd * 4 + c] * att_src[hd * 4 + c];
            dd += acc[hd * 4 + c] * att_dst[hd * 4 + c];
        }
        a_src[(size_t)r * 4 + hd] = s;
        a_dst[(size_t)r * 4 + hd] = dd;
        emax [(size_t)r * 4 + hd] = 0u;   // below fenc of any finite float
        denom[(size_t)r * 4 + hd] = 0.f;
    }
#pragma unroll
    for (int c = 0; c < 16; ++c) agg[(size_t)r * 16 + c] = 0.f;
}

// =====================================================================
// Kernel 2: per-edge segment max (includes self loops).
// =====================================================================
__global__ void edge_max_kernel(const int* __restrict__ eidx,
                                const float* __restrict__ a_src,
                                const float* __restrict__ a_dst,
                                unsigned* __restrict__ emax) {
    int e = blockIdx.x * blockDim.x + threadIdx.x;
    if (e >= ETOT) return;
    int s, d;
    if (e < NEDGES) { s = eidx[e]; d = eidx[NEDGES + e]; }
    else            { s = d = e - NEDGES; }
#pragma unroll
    for (int hd = 0; hd < NHEADS; ++hd) {
        float v = a_src[(size_t)s * 4 + hd] + a_dst[(size_t)d * 4 + hd];
        v = v > 0.f ? v : 0.2f * v;   // leaky relu
        atomicMax(&emax[(size_t)d * 4 + hd], fenc(v));
    }
}

// =====================================================================
// Kernel 3: per-edge exp / denom / unnormalized message aggregation.
// =====================================================================
__global__ void edge_sum_kernel(const int* __restrict__ eidx,
                                const float* __restrict__ a_src,
                                const float* __restrict__ a_dst,
                                const unsigned* __restrict__ emax,
                                const float* __restrict__ h,
                                float* __restrict__ denom,
                                float* __restrict__ agg) {
    int e = blockIdx.x * blockDim.x + threadIdx.x;
    if (e >= ETOT) return;
    int s, d;
    if (e < NEDGES) { s = eidx[e]; d = eidx[NEDGES + e]; }
    else            { s = d = e - NEDGES; }
#pragma unroll
    for (int hd = 0; hd < NHEADS; ++hd) {
        float v = a_src[(size_t)s * 4 + hd] + a_dst[(size_t)d * 4 + hd];
        v = v > 0.f ? v : 0.2f * v;
        float ee = __expf(v - fdec(emax[(size_t)d * 4 + hd]));
        atomicAdd(&denom[(size_t)d * 4 + hd], ee);
        const float* hs = h + (size_t)s * 16 + hd * 4;
#pragma unroll
        for (int c = 0; c < NEXP; ++c)
            atomicAdd(&agg[(size_t)d * 16 + hd * 4 + c], ee * hs[c]);
    }
}

// =====================================================================
// Kernel 4: normalize, +b_gat, layernorm(16), head mean, top-2 softmax.
// =====================================================================
__global__ void score_kernel(const float* __restrict__ agg,
                             const float* __restrict__ denom,
                             const float* __restrict__ b_gat,
                             const float* __restrict__ gamma,
                             const float* __restrict__ beta,
                             float* __restrict__ g) {
    int r = blockIdx.x * blockDim.x + threadIdx.x;
    if (r >= M_TOKENS) return;
    float a[16];
#pragma unroll
    for (int hd = 0; hd < NHEADS; ++hd) {
        float dn = denom[(size_t)r * 4 + hd] + 1e-16f;
#pragma unroll
        for (int c = 0; c < NEXP; ++c)
            a[hd * 4 + c] = agg[(size_t)r * 16 + hd * 4 + c] / dn + b_gat[hd * 4 + c];
    }
    float mu = 0.f;
#pragma unroll
    for (int i = 0; i < 16; ++i) mu += a[i];
    mu *= (1.f / 16.f);
    float var = 0.f;
#pragma unroll
    for (int i = 0; i < 16; ++i) { float t = a[i] - mu; var += t * t; }
    var *= (1.f / 16.f);
    float inv = rsqrtf(var + 1e-5f);
    float s[NEXP];
#pragma unroll
    for (int c = 0; c < NEXP; ++c) {
        float acc = 0.f;
#pragma unroll
        for (int hd = 0; hd < NHEADS; ++hd) {
            int i = hd * 4 + c;
            acc += (a[i] - mu) * inv * gamma[i] + beta[i];
        }
        s[c] = acc * 0.25f;
    }
    int i0 = 0;
#pragma unroll
    for (int c = 1; c < NEXP; ++c) if (s[c] > s[i0]) i0 = c;
    int i1 = -1;
#pragma unroll
    for (int c = 0; c < NEXP; ++c)
        if (c != i0 && (i1 < 0 || s[c] > s[i1])) i1 = c;
    float e1 = __expf(s[i1] - s[i0]);
    float w0 = 1.f / (1.f + e1);
    float w1 = e1 * w0;
#pragma unroll
    for (int c = 0; c < NEXP; ++c)
        g[(size_t)r * 4 + c] = (c == i0) ? w0 : ((c == i1) ? w1 : 0.f);
}

// =====================================================================
// Kernel 5: fused dense MoE via bf16 WMMA + TDM-staged activations.
//   per block: 32-token tile; bf16 x tile DMA'd into LDS by the Tensor
//   Data Mover (pad_interval/pad_amount reproduce the 264-element pitch);
//   per expert: GEMM1 -> exact GELU -> bf16 hh in LDS -> GEMM2 ->
//   out_acc += g * (tmp + b2); final dense store.
// =====================================================================
#define MT  32
#define LDX 264   // 256 + 8 bf16 pad = 128 + 4 DWORDs -> TDM pad_interval=6, pad_amount=3
#define LDH 520   // 512 + 8 bf16 pad (row pitch 1040B, 16B aligned)

__global__ __launch_bounds__(256, 1)
void moe_kernel(const __bf16* __restrict__ xb,
                const __bf16* __restrict__ W1t, const float* __restrict__ b1,
                const __bf16* __restrict__ W2t, const float* __restrict__ b2,
                const float* __restrict__ g,  float* __restrict__ out) {
    __shared__ __align__(16) __bf16 sX[MT * LDX];
    __shared__ __align__(16) __bf16 sH[MT * LDH];
    __shared__ float sG[MT * NEXP];

    const int tid   = threadIdx.x;
    const int wave  = tid >> 5;
    const int lane  = tid & 31;
    const int lhalf = lane >> 4;   // K-interleave half / M half
    const int lmod  = lane & 15;   // row (A) or column (B, C/D)
    const int row0  = blockIdx.x * MT;

    // ---- TDM: async tensor load of the 32x256 bf16 x tile into LDS ----
    if (wave == 0) {
        unsigned long long ga = (unsigned long long)(uintptr_t)(xb + (size_t)row0 * D_DIM);
        unsigned ldsa = (unsigned)(uintptr_t)(&sX[0]);   // low 32 bits = LDS offset
        v4u g0;
        g0[0] = 1u;                                            // count=1, user mode
        g0[1] = ldsa;                                          // lds_addr
        g0[2] = (unsigned)(ga & 0xFFFFFFFFull);                // global_addr[31:0]
        g0[3] = (unsigned)((ga >> 32) & 0x1FFFFFFull)          // global_addr[56:32]
              | (2u << 30);                                    // type=2 (image)
        v8i g1;
        g1[0] = (1 << 16)            // data_size = 1 (2-byte elements)
              | (1 << 20)            // pad_enable
              | (6 << 22)            // pad_interval: every 128 DWORDs (one 256-elem row)
              | (3 << 25);           // pad_amount: 4 DWORDs (8 bf16) -> pitch 264
        g1[1] = (int)(256u << 16);   // tensor_dim0 = 256 (low 16)
        g1[2] = (int)(32u  << 16);   // tensor_dim0 hi=0 | tensor_dim1 = 32 (low 16)
        g1[3] = (int)(256u << 16);   // tensor_dim1 hi=0 | tile_dim0 = 256
        g1[4] = 32;                  // tile_dim1 = 32, tile_dim2 = 0
        g1[5] = 256;                 // tensor_dim0_stride = 256 elements
        g1[6] = 0;
        g1[7] = 0;
        v4i z4; z4[0] = 0; z4[1] = 0; z4[2] = 0; z4[3] = 0;
        v8i z8;
#pragma unroll
        for (int j = 0; j < 8; ++j) z8[j] = 0;
        __builtin_amdgcn_tensor_load_to_lds(g0, g1, z4, z4, z8, 0);
        __builtin_amdgcn_s_wait_tensorcnt(0);
    }
    for (int i = tid; i < MT * NEXP; i += 256)
        sG[i] = g[(size_t)row0 * NEXP + i];
    __syncthreads();

    v8f oacc[2][2];
#pragma unroll
    for (int rt = 0; rt < 2; ++rt)
#pragma unroll
        for (int ct = 0; ct < 2; ++ct)
#pragma unroll
            for (int v = 0; v < 8; ++v) oacc[rt][ct][v] = 0.f;

    for (int ei = 0; ei < NEXP; ++ei) {
        const __bf16* W1e = W1t + (size_t)ei * D_DIM * D2;  // [n<512][k<256]
        const float*  b1e = b1  + ei * D2;
        const __bf16* W2e = W2t + (size_t)ei * D2 * D_DIM;  // [n<256][k<512]
        const float*  b2e = b2  + ei * D_DIM;

        // ---- GEMM1: hh[32,512] = gelu(xtile @ W1 + b1), wave owns 64 cols
#pragma unroll
        for (int ct = 0; ct < 4; ++ct) {
            const int col = wave * 64 + ct * 16 + lmod;
            v8f acc[2];
#pragma unroll
            for (int rt = 0; rt < 2; ++rt)
#pragma unroll
                for (int v = 0; v < 8; ++v) acc[rt][v] = 0.f;
            for (int ks = 0; ks < D_DIM / 32; ++ks) {
                // B fragment: lane col, halves j -> K = klo + j (contiguous)
                const int klo = ks * 32 + lhalf * 16;
                const __bf16* bp = W1e + (size_t)col * D_DIM + klo;
                v8bf b0 = *(const v8bf*)bp;
                v8bf b1v = *(const v8bf*)(bp + 8);
                v16bf bf;
#pragma unroll
                for (int j = 0; j < 8; ++j) { bf[j] = b0[j]; bf[j + 8] = b1v[j]; }
#pragma unroll
                for (int rt = 0; rt < 2; ++rt) {
                    const __bf16* ap = &sX[(rt * 16 + lmod) * LDX + ks * 32 + lhalf * 16];
                    v8bf lo = *(const v8bf*)ap;
                    v8bf hi = *(const v8bf*)(ap + 8);
                    v16bf af;
#pragma unroll
                    for (int j = 0; j < 8; ++j) { af[j] = lo[j]; af[j + 8] = hi[j]; }
                    acc[rt] = __builtin_amdgcn_wmma_f32_16x16x32_bf16(
                        false, af, false, bf, (short)0, acc[rt], false, false);
                }
            }
            const float bias = b1e[col];
#pragma unroll
            for (int rt = 0; rt < 2; ++rt)
#pragma unroll
                for (int v = 0; v < 8; ++v) {
                    int rr = rt * 16 + lhalf * 8 + v;
                    float val = acc[rt][v] + bias;
                    val = 0.5f * val * (1.0f + erff(val * 0.70710678118654752f));
                    sH[rr * LDH + col] = (__bf16)val;
                }
        }
        __syncthreads();

        // ---- GEMM2: tmp[32,256] = hh @ W2, wave owns 32 cols
#pragma unroll
        for (int ct = 0; ct < 2; ++ct) {
            const int col = wave * 32 + ct * 16 + lmod;
            v8f acc[2];
#pragma unroll
            for (int rt = 0; rt < 2; ++rt)
#pragma unroll
                for (int v = 0; v < 8; ++v) acc[rt][v] = 0.f;
            for (int ks = 0; ks < D2 / 32; ++ks) {
                const int klo = ks * 32 + lhalf * 16;
                const __bf16* bp = W2e + (size_t)col * D2 + klo;
                v8bf b0 = *(const v8bf*)bp;
                v8bf b1v = *(const v8bf*)(bp + 8);
                v16bf bf;
#pragma unroll
                for (int j = 0; j < 8; ++j) { bf[j] = b0[j]; bf[j + 8] = b1v[j]; }
#pragma unroll
                for (int rt = 0; rt < 2; ++rt) {
                    const __bf16* ap = &sH[(rt * 16 + lmod) * LDH + ks * 32 + lhalf * 16];
                    v8bf lo = *(const v8bf*)ap;
                    v8bf hi = *(const v8bf*)(ap + 8);
                    v16bf af;
#pragma unroll
                    for (int j = 0; j < 8; ++j) { af[j] = lo[j]; af[j + 8] = hi[j]; }
                    acc[rt] = __builtin_amdgcn_wmma_f32_16x16x32_bf16(
                        false, af, false, bf, (short)0, acc[rt], false, false);
                }
            }
            const float bias2 = b2e[col];
#pragma unroll
            for (int rt = 0; rt < 2; ++rt)
#pragma unroll
                for (int v = 0; v < 8; ++v) {
                    int rr = rt * 16 + lhalf * 8 + v;
                    oacc[rt][ct][v] += sG[rr * NEXP + ei] * (acc[rt][v] + bias2);
                }
        }
        __syncthreads();   // sH reused next expert
    }

    // dense store of out tile
#pragma unroll
    for (int ct = 0; ct < 2; ++ct) {
        const int col = wave * 32 + ct * 16 + lmod;
#pragma unroll
        for (int rt = 0; rt < 2; ++rt)
#pragma unroll
            for (int v = 0; v < 8; ++v) {
                int rr = rt * 16 + lhalf * 8 + v;
                out[(size_t)(row0 + rr) * D_DIM + col] = oacc[rt][ct][v];
            }
    }
}

// =====================================================================
extern "C" void kernel_launch(void* const* d_in, const int* in_sizes, int n_in,
                              void* d_out, int out_size, void* d_ws, size_t ws_size,
                              hipStream_t stream) {
    const float* x       = (const float*)d_in[0];
    const float* W_gat   = (const float*)d_in[1];
    const float* att_src = (const float*)d_in[2];
    const float* att_dst = (const float*)d_in[3];
    const float* b_gat   = (const float*)d_in[4];
    const float* ln_g    = (const float*)d_in[5];
    const float* ln_b    = (const float*)d_in[6];
    const float* W1      = (const float*)d_in[7];
    const float* b1      = (const float*)d_in[8];
    const float* W2      = (const float*)d_in[9];
    const float* b2      = (const float*)d_in[10];
    const int*   eidx    = (const int*)d_in[11];
    float* out = (float*)d_out;

    // workspace carve-out:
    //   fp32/uint32 gating arrays: M*52*4B = 13.3MB
    //   bf16 staging: xb 32MB, W1t 1MB, W2t 1MB   -> total ~47.3MB
    float* h      = (float*)d_ws;
    float* a_src  = h      + (size_t)M_TOKENS * 16;
    float* a_dst  = a_src  + (size_t)M_TOKENS * 4;
    unsigned* emx = (unsigned*)(a_dst + (size_t)M_TOKENS * 4);
    float* denom  = (float*)(emx + (size_t)M_TOKENS * 4);
    float* agg    = denom  + (size_t)M_TOKENS * 4;
    float* g      = agg    + (size_t)M_TOKENS * 16;
    __bf16* xb    = (__bf16*)(g + (size_t)M_TOKENS * 4);
    __bf16* W1t   = xb  + (size_t)M_TOKENS * D_DIM;
    __bf16* W2t   = W1t + (size_t)NEXP * D_DIM * D2;

    convert_kernel<<<(M_TOKENS * D_DIM) / 256, 256, 0, stream>>>(
        x, W1, W2, xb, W1t, W2t);
    gating_kernel<<<M_TOKENS / 256, 256, 0, stream>>>(
        x, W_gat, att_src, att_dst, h, a_src, a_dst, emx, denom, agg);
    edge_max_kernel<<<(ETOT + 255) / 256, 256, 0, stream>>>(
        eidx, a_src, a_dst, emx);
    edge_sum_kernel<<<(ETOT + 255) / 256, 256, 0, stream>>>(
        eidx, a_src, a_dst, emx, h, denom, agg);
    score_kernel<<<M_TOKENS / 256, 256, 0, stream>>>(
        agg, denom, b_gat, ln_g, ln_b, g);
    moe_kernel<<<M_TOKENS / MT, 256, 0, stream>>>(
        xb, W1t, b1, W2t, b2, g, out);
}
